// PureRGCN_66778151518271
// MI455X (gfx1250) — matrix operand order repaired and checked by reading, not compiled, not run on hardware
//
#include <hip/hip_runtime.h>
#include <hip/hip_bf16.h>

// ---------------- problem constants ----------------
#define NN 30000      // nodes
#define NE 480000     // edges
#define NR 16         // relations
#define DD 128        // dim
#define NTRI 200000   // triples
#define MAXTILES 30016            // NE/16 + NR headroom for per-relation padding
#define SORTSZ (MAXTILES * 16)    // padded sorted-edge capacity

// ---------------- WMMA types ----------------
typedef __attribute__((ext_vector_type(16))) __bf16 v16bf;
typedef __attribute__((ext_vector_type(8)))  float  v8f;
typedef __attribute__((ext_vector_type(4)))  unsigned int u32x4;

union V16 { u32x4 q[2]; v16bf v; };

__device__ __forceinline__ unsigned short f2bf(float f) {
  unsigned int u = __float_as_uint(f);
  u += 0x7FFFu + ((u >> 16) & 1u);            // round-to-nearest-even
  return (unsigned short)(u >> 16);
}

// ---------------- bookkeeping kernels ----------------
__global__ void k_zero_meta(int* counts, int* curs, int* deg) {
  int i = blockIdx.x * blockDim.x + threadIdx.x;
  if (i < NR) { counts[i] = 0; curs[i] = 0; }
  if (i < NN) deg[i] = 0;
}

__global__ void k_hist(const int* __restrict__ etype, const int* __restrict__ dst,
                       int* counts, int* deg) {
  int i = blockIdx.x * blockDim.x + threadIdx.x;
  if (i >= NE) return;
  atomicAdd(&counts[etype[i]], 1);
  atomicAdd(&deg[dst[i]], 1);
}

// single block: per-relation padded bases, tile->relation map, sentinel pads
__global__ void k_setup(const int* __restrict__ counts, int* basep,
                        int* tile_rel, int* ssrc, int* sdst) {
  __shared__ int s_tb[NR + 1];
  __shared__ int s_bp[NR];
  __shared__ int s_cnt[NR];
  if (threadIdx.x == 0) {
    int run = 0, trun = 0;
    for (int r = 0; r < NR; ++r) {
      int c = counts[r];
      s_cnt[r] = c; s_bp[r] = run; basep[r] = run; s_tb[r] = trun;
      int tr = (c + 15) >> 4;
      trun += tr; run += tr << 4;
    }
    s_tb[NR] = trun;
  }
  __syncthreads();
  int total = s_tb[NR];
  for (int t = threadIdx.x; t < MAXTILES; t += blockDim.x) {
    int rr = -1;
    if (t < total)
      for (int r = 0; r < NR; ++r)
        if (t >= s_tb[r] && t < s_tb[r + 1]) { rr = r; break; }
    tile_rel[t] = rr;
  }
  if (threadIdx.x < NR) {
    int r = threadIdx.x;
    int end = s_bp[r] + (((s_cnt[r] + 15) >> 4) << 4);
    for (int p = s_bp[r] + s_cnt[r]; p < end; ++p) { ssrc[p] = 0; sdst[p] = -1; }
  }
}

__global__ void k_scatter(const int* __restrict__ src, const int* __restrict__ dst,
                          const int* __restrict__ etype, const int* __restrict__ basep,
                          int* curs, int* ssrc, int* sdst) {
  int i = blockIdx.x * blockDim.x + threadIdx.x;
  if (i >= NE) return;
  int r = etype[i];
  int pos = basep[r] + atomicAdd(&curs[r], 1);
  ssrc[pos] = src[i];
  sdst[pos] = dst[i];
}

// ---------------- packing / conversion ----------------
// W_rel[r][k][n] (in,out) -> packed[r][n][k] bf16 so B K-values are contiguous
__global__ void k_pack_rel(const float* __restrict__ W, unsigned short* __restrict__ out) {
  int i = blockIdx.x * blockDim.x + threadIdx.x;
  if (i >= NR * DD * DD) return;
  int r = i >> 14, rem = i & 16383, k = rem >> 7, n = rem & 127;
  out[(r << 14) + (n << 7) + k] = f2bf(W[i]);
}

// Wself[out][in] is already [n][k]: straight convert
__global__ void k_pack_mat(const float* __restrict__ W, unsigned short* __restrict__ out) {
  int i = blockIdx.x * blockDim.x + threadIdx.x;
  if (i >= DD * DD) return;
  out[i] = f2bf(W[i]);
}

__global__ void k_gather_x0(const float* __restrict__ emb, const int* __restrict__ ln,
                            float* __restrict__ x) {
  int i = blockIdx.x * blockDim.x + threadIdx.x;
  if (i >= NN * DD) return;
  int n = i >> 7, d = i & 127;
  x[i] = emb[(size_t)ln[n] * DD + d];
}

__global__ void k_tobf(const float* __restrict__ x, unsigned short* __restrict__ xbf) {
  int i = blockIdx.x * blockDim.x + threadIdx.x;
  if (i >= NN * DD) return;
  xbf[i] = f2bf(x[i]);
}

__global__ void k_zero_agg(float* agg) {
  int i = blockIdx.x * blockDim.x + threadIdx.x;
  if (i < NN * DD) agg[i] = 0.0f;
}

// ---------------- WMMA GEMM kernels ----------------
// self loop: one wave computes rows [tile*16, tile*16+16) x all 128 cols
__global__ void __launch_bounds__(32)
k_self_gemm(const unsigned short* __restrict__ xbf, const unsigned short* __restrict__ wsbf,
            const float* __restrict__ bias, float* __restrict__ out) {
  int tile = blockIdx.x;
  int lane = threadIdx.x;
  int m = lane & 15, half = lane >> 4;
  const unsigned short* xrow = xbf + (size_t)(tile * 16 + m) * DD;
  v8f acc[8];
#pragma unroll
  for (int i = 0; i < 8; ++i) acc[i] = (v8f){0.f,0.f,0.f,0.f,0.f,0.f,0.f,0.f};
#pragma unroll
  for (int kt = 0; kt < 4; ++kt) {
    int kk = kt * 32;
    V16 a;
    a.q[0] = *(const u32x4*)(xrow + kk + half * 16);       // K kk+half*16 .. +7
    a.q[1] = *(const u32x4*)(xrow + kk + half * 16 + 8);   // K kk+half*16+8 .. +15
#pragma unroll
    for (int nt = 0; nt < 8; ++nt) {
      int n = nt * 16 + m;
      const u32x4* bp = (const u32x4*)(wsbf + (size_t)n * DD + kk + half * 16);
      V16 b; b.q[0] = bp[0]; b.q[1] = bp[1];               // K 16 consecutive
      acc[nt] = __builtin_amdgcn_wmma_f32_16x16x32_bf16(
          false, a.v, false, b.v, (short)0, acc[nt], false, false);
    }
  }
#pragma unroll
  for (int j = 0; j < 8; ++j) {
    int node = tile * 16 + j + half * 8;
    float* op = out + (size_t)node * DD;
#pragma unroll
    for (int nt = 0; nt < 8; ++nt) {
      int n = nt * 16 + m;
      op[n] = acc[nt][j] + bias[n];
    }
  }
}

// edge messages: one wave per 16-edge tile of one relation; A gathered by src
__global__ void __launch_bounds__(32)
k_edge_gemm(const unsigned short* __restrict__ xbf, const unsigned short* __restrict__ wbf,
            const int* __restrict__ ssrc, const int* __restrict__ sdst,
            const int* __restrict__ tile_rel, float* __restrict__ agg) {
  int tile = blockIdx.x;
  int r = tile_rel[tile];
  if (r < 0) return;
  int lane = threadIdx.x;
  int m = lane & 15, half = lane >> 4;
  int src = ssrc[tile * 16 + m];
  const unsigned short* xrow = xbf + (size_t)src * DD;
  const unsigned short* wb = wbf + ((size_t)r << 14);
  v8f acc[8];
#pragma unroll
  for (int i = 0; i < 8; ++i) acc[i] = (v8f){0.f,0.f,0.f,0.f,0.f,0.f,0.f,0.f};
#pragma unroll
  for (int kt = 0; kt < 4; ++kt) {
    int kk = kt * 32;
    V16 a;
    a.q[0] = *(const u32x4*)(xrow + kk + half * 16);
    a.q[1] = *(const u32x4*)(xrow + kk + half * 16 + 8);
#pragma unroll
    for (int nt = 0; nt < 8; ++nt) {
      int n = nt * 16 + m;
      const u32x4* bp = (const u32x4*)(wb + (size_t)n * DD + kk + half * 16);
      V16 b; b.q[0] = bp[0]; b.q[1] = bp[1];
      acc[nt] = __builtin_amdgcn_wmma_f32_16x16x32_bf16(
          false, a.v, false, b.v, (short)0, acc[nt], false, false);
    }
  }
#pragma unroll
  for (int j = 0; j < 8; ++j) {
    int d = sdst[tile * 16 + j + half * 8];     // C row j+8*half
    if (d >= 0) {
      float* dp = agg + (size_t)d * DD;
#pragma unroll
      for (int nt = 0; nt < 8; ++nt)
        atomicAdd(dp + nt * 16 + m, acc[nt][j]);
    }
  }
}

__global__ void k_combine(const float* __restrict__ selfo, const float* __restrict__ agg,
                          const int* __restrict__ deg, float* __restrict__ out) {
  int i = blockIdx.x * blockDim.x + threadIdx.x;
  if (i >= NN * DD) return;
  float dg = fmaxf((float)deg[i >> 7], 1.0f);
  float v = selfo[i] + agg[i] / dg;
  out[i] = v > 0.0f ? v : 0.0f;
}

// DistMult: one wave per triple, float4 loads, butterfly reduce
__global__ void k_score(const float* __restrict__ x, const float* __restrict__ rel,
                        const int* __restrict__ h, const int* __restrict__ r,
                        const int* __restrict__ t, float* __restrict__ out) {
  int gid = blockIdx.x * blockDim.x + threadIdx.x;
  int w = gid >> 5, lane = gid & 31;
  if (w >= NTRI) return;
  const float4* X = (const float4*)x;
  const float4* R = (const float4*)rel;
  float4 a = X[(size_t)h[w] * 32 + lane];
  float4 b = R[(size_t)r[w] * 32 + lane];
  float4 c = X[(size_t)t[w] * 32 + lane];
  float s = a.x * b.x * c.x + a.y * b.y * c.y + a.z * b.z * c.z + a.w * b.w * c.w;
  for (int off = 16; off; off >>= 1) s += __shfl_xor(s, off, 32);
  if (lane == 0) out[w] = s;
}

// ---------------- launch ----------------
static constexpr size_t al256(size_t x) { return (x + 255) & ~(size_t)255; }

extern "C" void kernel_launch(void* const* d_in, const int* in_sizes, int n_in,
                              void* d_out, int out_size, void* d_ws, size_t ws_size,
                              hipStream_t stream) {
  const float* node_emb = (const float*)d_in[0];
  const float* rel_emb  = (const float*)d_in[1];
  const float* W_rel0   = (const float*)d_in[2];
  const float* Wself_w0 = (const float*)d_in[3];
  const float* Wself_b0 = (const float*)d_in[4];
  const float* W_rel1   = (const float*)d_in[5];
  const float* Wself_w1 = (const float*)d_in[6];
  const float* Wself_b1 = (const float*)d_in[7];
  const int*   local_nodes = (const int*)d_in[8];
  const int*   edge_index  = (const int*)d_in[9];   // [2, NE]
  const int*   edge_type   = (const int*)d_in[10];
  const int*   h_loc = (const int*)d_in[11];
  const int*   r_ids = (const int*)d_in[12];
  const int*   t_loc = (const int*)d_in[13];
  float* scores = (float*)d_out;

  const int* e_src = edge_index;
  const int* e_dst = edge_index + NE;

  char* w = (char*)d_ws;
  const size_t oCounts = 0;
  const size_t oCurs   = 256;
  const size_t oBase   = 512;
  const size_t oDeg    = 1024;
  const size_t oTiler  = oDeg   + al256((size_t)NN * 4);
  const size_t oSsrc   = oTiler + al256((size_t)MAXTILES * 4);
  const size_t oSdst   = oSsrc  + al256((size_t)SORTSZ * 4);
  const size_t oXbf    = oSdst  + al256((size_t)SORTSZ * 4);
  const size_t oWbf0   = oXbf   + al256((size_t)NN * DD * 2);
  const size_t oWbf1   = oWbf0  + (size_t)NR * DD * DD * 2;
  const size_t oWs0    = oWbf1  + (size_t)NR * DD * DD * 2;
  const size_t oWs1    = oWs0   + (size_t)DD * DD * 2;
  const size_t oXA     = al256(oWs1 + (size_t)DD * DD * 2);
  const size_t oXB     = oXA    + (size_t)NN * DD * 4;
  const size_t oSelf   = oXB    + (size_t)NN * DD * 4;
  const size_t oAgg    = oSelf  + (size_t)NN * DD * 4;

  int* counts  = (int*)(w + oCounts);
  int* curs    = (int*)(w + oCurs);
  int* basep   = (int*)(w + oBase);
  int* deg     = (int*)(w + oDeg);
  int* tilerel = (int*)(w + oTiler);
  int* ssrc    = (int*)(w + oSsrc);
  int* sdst    = (int*)(w + oSdst);
  unsigned short* xbf  = (unsigned short*)(w + oXbf);
  unsigned short* wbf0 = (unsigned short*)(w + oWbf0);
  unsigned short* wbf1 = (unsigned short*)(w + oWbf1);
  unsigned short* ws0  = (unsigned short*)(w + oWs0);
  unsigned short* ws1  = (unsigned short*)(w + oWs1);
  float* xA   = (float*)(w + oXA);
  float* xB   = (float*)(w + oXB);
  float* selfo = (float*)(w + oSelf);
  float* agg  = (float*)(w + oAgg);

  const int TB = 256;
  const int gElem  = (NN * DD + TB - 1) / TB;   // 15000
  const int gEdge  = (NE + TB - 1) / TB;        // 1875
  const int gNode  = (NN + TB - 1) / TB;
  const int gTiles = NN / 16;                   // 1875 self-loop tiles

  // ---- sort edges by relation + degree ----
  k_zero_meta<<<gNode, TB, 0, stream>>>(counts, curs, deg);
  k_hist<<<gEdge, TB, 0, stream>>>(edge_type, e_dst, counts, deg);
  k_setup<<<1, 256, 0, stream>>>(counts, basep, tilerel, ssrc, sdst);
  k_scatter<<<gEdge, TB, 0, stream>>>(e_src, e_dst, edge_type, basep, curs, ssrc, sdst);

  // ---- pack weights to bf16 ----
  k_pack_rel<<<(NR * DD * DD + TB - 1) / TB, TB, 0, stream>>>(W_rel0, wbf0);
  k_pack_rel<<<(NR * DD * DD + TB - 1) / TB, TB, 0, stream>>>(W_rel1, wbf1);
  k_pack_mat<<<(DD * DD + TB - 1) / TB, TB, 0, stream>>>(Wself_w0, ws0);
  k_pack_mat<<<(DD * DD + TB - 1) / TB, TB, 0, stream>>>(Wself_w1, ws1);

  // ---- x0 = node_emb[local_nodes] ----
  k_gather_x0<<<gElem, TB, 0, stream>>>(node_emb, local_nodes, xA);

  // ---- layer 0: xA -> xB ----
  k_tobf<<<gElem, TB, 0, stream>>>(xA, xbf);
  k_self_gemm<<<gTiles, 32, 0, stream>>>(xbf, ws0, Wself_b0, selfo);
  k_zero_agg<<<gElem, TB, 0, stream>>>(agg);
  k_edge_gemm<<<MAXTILES, 32, 0, stream>>>(xbf, wbf0, ssrc, sdst, tilerel, agg);
  k_combine<<<gElem, TB, 0, stream>>>(selfo, agg, deg, xB);

  // ---- layer 1: xB -> xA ----
  k_tobf<<<gElem, TB, 0, stream>>>(xB, xbf);
  k_self_gemm<<<gTiles, 32, 0, stream>>>(xbf, ws1, Wself_b1, selfo);
  k_zero_agg<<<gElem, TB, 0, stream>>>(agg);
  k_edge_gemm<<<MAXTILES, 32, 0, stream>>>(xbf, wbf1, ssrc, sdst, tilerel, agg);
  k_combine<<<gElem, TB, 0, stream>>>(selfo, agg, deg, xA);

  // ---- DistMult scoring ----
  k_score<<<(NTRI * 32 + TB - 1) / TB, TB, 0, stream>>>(xA, rel_emb, h_loc, r_ids, t_loc, scores);
}
